// QuantDeformConvWithOffsetScaleBoundPositive_70901320123225
// MI455X (gfx1250) — matrix operand order repaired and checked by reading, not compiled, not run on hardware
//
#include <hip/hip_runtime.h>
#include <hip/hip_bf16.h>
#include <stdint.h>

// Problem constants (match reference)
#define B_     8
#define CIN    256
#define COUT   256
#define H_     64
#define W_     64
#define HW     4096          // H*W
#define NPOS   32768         // B*H*W
#define KTOT   2304          // 9*CIN
#define NKC    72            // KTOT/32 k-chunks
#define NT     32            // N-tile (pixels per block) = 2 WMMA n-tiles
#define BSTRIDE 264          // LDS row stride (bf16 elems), padded: 528B -> 132 dwords -> bank step 4

typedef __attribute__((ext_vector_type(16))) __bf16 v16bf;
typedef __attribute__((ext_vector_type(8)))  float  v8f;

__device__ __forceinline__ unsigned short f32_to_bf16(float f) {
  union { float f; unsigned int u; } v; v.f = f;
  unsigned int r = v.u + 0x7FFFu + ((v.u >> 16) & 1u);   // RNE
  return (unsigned short)(r >> 16);
}

__device__ __forceinline__ float quantq(float w, float s) {
  float q = rintf(w / s);                 // round-half-even like jnp.round
  q = fminf(fmaxf(q, -8.0f), 7.0f);       // clip(-n-1, n), n=7
  return q * s;
}

// ---------------- kernel 1: per-tensor min/max -> scale s ----------------
__global__ void minmax_kernel(const float* __restrict__ w_off,
                              const float* __restrict__ w_scale,
                              const float* __restrict__ w_dcn,
                              float* __restrict__ sbuf) {
  const float* ptr; int n;
  if (blockIdx.x == 0)      { ptr = w_off;   n = 2 * CIN * 9; }
  else if (blockIdx.x == 1) { ptr = w_scale; n = CIN * 9; }
  else                      { ptr = w_dcn;   n = COUT * CIN * 9; }
  float mn = 3.4e38f, mx = -3.4e38f;
  for (int i = threadIdx.x; i < n; i += 256) {
    float v = ptr[i];
    mn = fminf(mn, v); mx = fmaxf(mx, v);
  }
  __shared__ float smn[256], smx[256];
  smn[threadIdx.x] = mn; smx[threadIdx.x] = mx;
  __syncthreads();
  for (int s = 128; s > 0; s >>= 1) {
    if (threadIdx.x < s) {
      smn[threadIdx.x] = fminf(smn[threadIdx.x], smn[threadIdx.x + s]);
      smx[threadIdx.x] = fmaxf(smx[threadIdx.x], smx[threadIdx.x + s]);
    }
    __syncthreads();
  }
  if (threadIdx.x == 0) {
    float mag = fmaxf(fabsf(smn[0]), fabsf(smx[0]));
    sbuf[blockIdx.x] = fmaxf(mag, 1e-8f) / 7.0f;
  }
}

// ---------------- kernel 2: quantize offset/scale conv weights ----------------
__global__ void quant_small_kernel(const float* __restrict__ w_off,
                                   const float* __restrict__ w_scale,
                                   const float* __restrict__ sbuf,
                                   float* __restrict__ woq,
                                   float* __restrict__ wsq) {
  int id = blockIdx.x * 256 + threadIdx.x;
  if (id < 2 * CIN * 9) woq[id] = quantq(w_off[id], sbuf[0]);
  int id2 = id - 2 * CIN * 9;
  if (id2 >= 0 && id2 < CIN * 9) wsq[id2] = quantq(w_scale[id2], sbuf[1]);
}

// ---------------- kernel 3: quantize + pack w_dcn into WMMA A-fragment layout ----------------
// Layout: apack[((mtile*NKC + kc)*32 + lane)*16 + j], bf16.
// A 16x32 in this code's contiguous K order (B matches): lane<16 -> M=lane, K in {0..15};
// lane>=16 -> same M, K in {16..31}.  K(j,lane)= ((lane>>4)<<4)+j
__global__ void pack_wdcn_kernel(const float* __restrict__ w_dcn,
                                 const float* __restrict__ sbuf,
                                 unsigned short* __restrict__ apack) {
  int flat = blockIdx.x * 256 + threadIdx.x;      // < 16*72*32*16 = 589824
  float s = sbuf[2];
  int j     = flat & 15;
  int lane  = (flat >> 4) & 31;
  int t3    = flat >> 9;
  int kcg   = t3 % NKC;
  int mtile = t3 / NKC;
  int m  = mtile * 16 + (lane & 15);
  int kl = ((lane >> 4) << 4) + j;
  int K  = kcg * 32 + kl;                          // global K = anchor*256 + c
  int k9 = K >> 8, c = K & 255;
  float w = w_dcn[(m * CIN + c) * 9 + k9];         // [COUT,CIN,3,3] row-major
  apack[flat] = f32_to_bf16(quantq(w, s));
}

// ---------------- kernel 4: offset / scale maps (3-channel conv3x3, pad 1) ----------------
__global__ void __launch_bounds__(256)
offset_scale_kernel(const float* __restrict__ x,
                    const float* __restrict__ woq,
                    const float* __restrict__ wsq,
                    const float* __restrict__ b_off,
                    const float* __restrict__ b_scale,
                    float* __restrict__ offmap) {
  __shared__ float sw[3 * CIN * 9];                // 27.6 KB
  for (int i = threadIdx.x; i < 2 * CIN * 9; i += 256) sw[i] = woq[i];
  for (int i = threadIdx.x; i < CIN * 9; i += 256) sw[2 * CIN * 9 + i] = wsq[i];
  __syncthreads();

  int pos = blockIdx.x * 256 + threadIdx.x;        // < NPOS
  int b = pos >> 12, h = (pos >> 6) & 63, w = pos & 63;
  const float* xb = x + (size_t)b * CIN * HW;
  float a0 = 0.f, a1 = 0.f, a2 = 0.f;
  for (int c = 0; c < CIN; ++c) {
    const float* xc = xb + c * HW;
#pragma unroll
    for (int dy = 0; dy < 3; ++dy) {
      int y = h + dy - 1;
      if (y < 0 || y >= H_) continue;
#pragma unroll
      for (int dx = 0; dx < 3; ++dx) {
        int xw = w + dx - 1;
        if (xw < 0 || xw >= W_) continue;
        float v = xc[y * W_ + xw];
        int ko = c * 9 + dy * 3 + dx;
        a0 = fmaf(v, sw[ko], a0);                  // w_off[0]
        a1 = fmaf(v, sw[CIN * 9 + ko], a1);        // w_off[1]
        a2 = fmaf(v, sw[2 * CIN * 9 + ko], a2);    // w_scale[0]
      }
    }
  }
  a0 += b_off[0];
  a1 += b_off[1];
  a2 = fmaxf(a2 + b_scale[0], 0.0f);               // relu
  offmap[pos] = a0;
  offmap[NPOS + pos] = a1;
  offmap[2 * NPOS + pos] = a2;
}

// ---------------- kernel 5: deformable conv as bf16 WMMA GEMM ----------------
// Block: 256 threads (8 waves). Tile: N=32 pixels x M=256 channels.
// Wave w owns M rows [w*32, w*32+32) x both 16-pixel n-tiles -> 2x2 accumulators,
// so every A fragment and every B fragment feeds two WMMAs.
__global__ void __launch_bounds__(256)
dcn_wmma_kernel(const float* __restrict__ x,
                const unsigned short* __restrict__ apack,
                const float* __restrict__ offmap,
                const float* __restrict__ b_dcn,
                float* __restrict__ out) {
  __shared__ int   sO00[NT * 9], sO01[NT * 9], sO10[NT * 9], sO11[NT * 9];
  __shared__ float sW00[NT * 9], sW01[NT * 9], sW10[NT * 9], sW11[NT * 9];
  __shared__ __align__(32) unsigned short smem_b[NT * BSTRIDE];   // 16.9 KB

  const int tid  = threadIdx.x;
  const int lane = tid & 31;
  const int wid  = tid >> 5;
  const int pos0 = blockIdx.x * NT;                // 32 consecutive pixels: same b,h
  const int b  = pos0 >> 12;
  const int h  = (pos0 >> 6) & 63;
  const int w0 = pos0 & 63;
  const float* xb = x + (size_t)b * CIN * HW;

  // Per-(pixel, anchor) bilinear params; invalid corners -> weight 0 (zero pad).
  for (int t = tid; t < NT * 9; t += 256) {
    int n = t / 9, k = t % 9;
    int pos = pos0 + n;
    float o0 = offmap[pos];
    float o1 = offmap[NPOS + pos];
    float sc = offmap[2 * NPOS + pos];
    float ryf = (float)(k / 3 - 1), rxf = (float)(k % 3 - 1);
    float py = (float)h + ryf * sc + o0;
    float px = (float)(w0 + n) + rxf * sc + o1;
    float y0f = floorf(py), x0f = floorf(px);
    float wy = py - y0f, wx = px - x0f;
    int y0 = (int)y0f, x0i = (int)x0f;
    int y1 = y0 + 1, x1i = x0i + 1;
    bool vy0 = (y0 >= 0) && (y0 < H_), vy1 = (y1 >= 0) && (y1 < H_);
    bool vx0 = (x0i >= 0) && (x0i < W_), vx1 = (x1i >= 0) && (x1i < W_);
    int y0c = min(max(y0, 0), H_ - 1), y1c = min(max(y1, 0), H_ - 1);
    int x0c = min(max(x0i, 0), W_ - 1), x1c = min(max(x1i, 0), W_ - 1);
    sO00[t] = y0c * W_ + x0c;
    sO01[t] = y0c * W_ + x1c;
    sO10[t] = y1c * W_ + x0c;
    sO11[t] = y1c * W_ + x1c;
    sW00[t] = (vy0 && vx0) ? (1.f - wy) * (1.f - wx) : 0.f;
    sW01[t] = (vy0 && vx1) ? (1.f - wy) * wx : 0.f;
    sW10[t] = (vy1 && vx0) ? wy * (1.f - wx) : 0.f;
    sW11[t] = (vy1 && vx1) ? wy * wx : 0.f;
  }

  v8f acc00 = {}, acc01 = {}, acc10 = {}, acc11 = {};
  const int mt0 = wid * 2;                         // first of 2 m-tiles for this wave

  __syncthreads();

  for (int k = 0; k < 9; ++k) {
    // ---- produce: bilinear-sample 32 pixels x 256 channels of anchor k -> LDS bf16 ----
    // Each thread owns one pixel row n and channel pairs (2c,2c+1) -> packed b32 stores.
    {
      int n = tid & 31;
      int cb = tid >> 5;                           // 0..7
      int pk = n * 9 + k;
      int o00 = sO00[pk], o01 = sO01[pk], o10 = sO10[pk], o11 = sO11[pk];
      float W00 = sW00[pk], W01 = sW01[pk], W10 = sW10[pk], W11 = sW11[pk];
      unsigned short* dst = &smem_b[n * BSTRIDE];
#pragma unroll
      for (int i = 0; i < 16; ++i) {
        int c0 = ((i << 3) + cb) << 1;             // even channel, covers 0..254
        const float* xc0 = xb + c0 * HW;
        const float* xc1 = xc0 + HW;
        float v0 = W00 * xc0[o00] + W01 * xc0[o01] + W10 * xc0[o10] + W11 * xc0[o11];
        float v1 = W00 * xc1[o00] + W01 * xc1[o01] + W10 * xc1[o10] + W11 * xc1[o11];
        unsigned int p = (unsigned int)f32_to_bf16(v0)
                       | ((unsigned int)f32_to_bf16(v1) << 16);
        *(unsigned int*)(dst + c0) = p;            // 4B-aligned: c0 even, row 528B
      }
    }
    __syncthreads();

    // ---- consume: 8 k-chunks of 32; 2 A frags x 2 B frags -> 4 WMMA per chunk ----
    // B fragment (32x16 K x N): lane<16 -> N=lane, K=chunk+0..15; lane>=16 -> K=chunk+16..31.
    const unsigned short* b0base =
        &smem_b[(lane & 15) * BSTRIDE + ((lane >> 4) << 4)];
    const unsigned short* b1base = b0base + 16 * BSTRIDE;   // second n-tile
#pragma unroll
    for (int kc = 0; kc < 8; ++kc) {
      v16bf bf0 = *(const v16bf*)(b0base + kc * 32);
      v16bf bf1 = *(const v16bf*)(b1base + kc * 32);
      int kcg = k * 8 + kc;
      const unsigned short* ap =
          apack + (((size_t)mt0 * NKC + kcg) * 32 + lane) * 16;
      v16bf a0 = *(const v16bf*)ap;
      v16bf a1 = *(const v16bf*)(ap + (size_t)NKC * 32 * 16);  // next m-tile
      acc00 = __builtin_amdgcn_wmma_f32_16x16x32_bf16(
          false, a0, false, bf0, (short)0, acc00, false, false);
      acc01 = __builtin_amdgcn_wmma_f32_16x16x32_bf16(
          false, a0, false, bf1, (short)0, acc01, false, false);
      acc10 = __builtin_amdgcn_wmma_f32_16x16x32_bf16(
          false, a1, false, bf0, (short)0, acc10, false, false);
      acc11 = __builtin_amdgcn_wmma_f32_16x16x32_bf16(
          false, a1, false, bf1, (short)0, acc11, false, false);
    }
    __syncthreads();
  }

  // ---- epilogue: D layout (16x16 f32): lane<16 -> M=r, N=lane; lane>=16 -> M=8+r ----
  int nl = lane & 15;
  int mrow = (lane >> 4) * 8;
#pragma unroll
  for (int r = 0; r < 8; ++r) {
    int m0 = mt0 * 16 + mrow + r;
    int m1 = m0 + 16;
    float bb0 = b_dcn[m0], bb1 = b_dcn[m1];
    size_t base0 = ((size_t)b * COUT + m0) * HW + h * W_ + w0;
    size_t base1 = ((size_t)b * COUT + m1) * HW + h * W_ + w0;
    out[base0 + nl]      = acc00[r] + bb0;
    out[base0 + 16 + nl] = acc01[r] + bb0;
    out[base1 + nl]      = acc10[r] + bb1;
    out[base1 + 16 + nl] = acc11[r] + bb1;
  }
}

// ---------------- launcher ----------------
extern "C" void kernel_launch(void* const* d_in, const int* in_sizes, int n_in,
                              void* d_out, int out_size, void* d_ws, size_t ws_size,
                              hipStream_t stream) {
  (void)in_sizes; (void)n_in; (void)out_size; (void)ws_size;
  const float* x       = (const float*)d_in[0];
  const float* w_off   = (const float*)d_in[1];
  const float* b_off   = (const float*)d_in[2];
  const float* w_scale = (const float*)d_in[3];
  const float* b_scale = (const float*)d_in[4];
  const float* w_dcn   = (const float*)d_in[5];
  const float* b_dcn   = (const float*)d_in[6];
  float* out = (float*)d_out;

  // Workspace layout (bytes):
  //   0      : sbuf[3] scales (pad to 64)
  //   64     : woq   4608 f32
  //   18496  : wsq   2304 f32
  //   27712  : offmap 3*32768 f32
  //   420928 : apack 589824 bf16  (ends ~1.53 MB)
  char* ws = (char*)d_ws;
  float* sbuf   = (float*)(ws + 0);
  float* woq    = (float*)(ws + 64);
  float* wsq    = (float*)(ws + 64 + 4608 * 4);
  float* offmap = (float*)(ws + 64 + 4608 * 4 + 2304 * 4);
  unsigned short* apack =
      (unsigned short*)(ws + 64 + 4608 * 4 + 2304 * 4 + 3 * NPOS * 4);

  minmax_kernel<<<3, 256, 0, stream>>>(w_off, w_scale, w_dcn, sbuf);
  quant_small_kernel<<<27, 256, 0, stream>>>(w_off, w_scale, sbuf, woq, wsq);
  pack_wdcn_kernel<<<2304, 256, 0, stream>>>(w_dcn, sbuf, apack);
  offset_scale_kernel<<<128, 256, 0, stream>>>(x, woq, wsq, b_off, b_scale, offmap);
  dcn_wmma_kernel<<<1024, 256, 0, stream>>>(x, apack, offmap, b_dcn, out);
}
